// PPOStructuredInsertionModel_54168127537174
// MI455X (gfx1250) — matrix operand order repaired and checked by reading, not compiled, not run on hardware
//
#include <hip/hip_runtime.h>
#include <cstdint>

// ---------------------------------------------------------------------------
// PPO structured-insertion model for MI455X (gfx1250, wave32, WMMA).
// Three 3-layer MLPs over 65536 rows.
// GEMMs: f16 WMMA, f32 accumulate, double-buffered async global->LDS staging.
// ---------------------------------------------------------------------------

typedef __attribute__((ext_vector_type(16))) _Float16 v16h;
typedef __attribute__((ext_vector_type(8)))  _Float16 v8h;
typedef __attribute__((ext_vector_type(8)))  float    v8f;

#define M_ROWS     65536             // T*B
#define IN_DIM     512
#define OBS_STRIDE 515               // 512 core + 3 gate
#define H_DIM      1024
#define NA_DIM     256

constexpr int BM  = 128;   // block tile M
constexpr int BN  = 128;   // block tile N
constexpr int BK  = 32;    // K step == WMMA K
constexpr int BKP = 40;    // padded LDS row pitch (halfs); 80B keeps 16B align

// ---- CDNA5 async global->LDS copy (ASYNCcnt-tracked), per ISA 08_async ----
__device__ __forceinline__ void async_b128(const _Float16* g, uint32_t lds_off)
{
    asm volatile("global_load_async_to_lds_b128 %0, %1, off"
                 :: "v"(lds_off), "v"((uint64_t)(uintptr_t)g)
                 : "memory");
}
__device__ __forceinline__ void wait_async_le4()
{
    asm volatile("s_wait_asynccnt 0x4" ::: "memory");
}
__device__ __forceinline__ void wait_async_0()
{
    asm volatile("s_wait_asynccnt 0x0" ::: "memory");
}
__device__ __forceinline__ uint32_t lds_off_u32(const void* p)
{
    // flat pointers into the shared aperture carry the LDS byte address in
    // the low 32 bits (ISA 10.2)
    return (uint32_t)(uintptr_t)p;
}

// ---------------------------------------------------------------------------
// Tiled GEMM: Out[m,n] = act( sum_k A[m,k] * W[n,k] + bias[n] )
//   A: f16 [M x lda], W: f16 [N x K] row-major (pre-converted, L2-resident).
//   256 thr = 8 waves (4x2); wave tile 32x64 -> eight v_wmma_f32_16x16x32_f16
//   per K step. Double-buffered LDS fed by async-to-LDS DMA.
// ---------------------------------------------------------------------------
template <bool OUT_F16, bool TANH>
__global__ __launch_bounds__(256) void gemm_wmma_async_kernel(
    const _Float16* __restrict__ A, long lda,
    const _Float16* __restrict__ W,
    const float* __restrict__ bias,
    void* __restrict__ Out, long ldo,
    int K)
{
    __shared__ __align__(16) _Float16 As[2][BM][BKP];   // 2 x 10KB
    __shared__ __align__(16) _Float16 Bs[2][BN][BKP];   // 2 x 10KB

    const int tid  = threadIdx.x;
    const int wave = tid >> 5;         // 8 waves
    const int lane = tid & 31;
    const int wm   = wave >> 1;        // 0..3  (M)
    const int wn   = wave & 1;         // 0..1  (N)
    const int l15  = lane & 15;
    const int hi   = lane >> 4;
    const int kb8  = hi * 8;           // unused; A frag K base is kb16
    const int kb16 = hi * 16;          // B frag K base

    const long bm0 = (long)blockIdx.x * BM;
    const long bn0 = (long)blockIdx.y * BN;

    // staging: A/B tiles are 128x32 halfs = 512 16B-chunks each; 2 per thread
    const int c0 = tid * 2, c1 = c0 + 1;
    const int ar0 = c0 >> 2, ac0 = (c0 & 3) * 8;
    const int ar1 = c1 >> 2, ac1 = (c1 & 3) * 8;

    v8f acc[2][4] = {};

    auto issue_tile = [&](int p, int k0) {
        async_b128(A + (bm0 + ar0) * lda + k0 + ac0, lds_off_u32(&As[p][ar0][ac0]));
        async_b128(A + (bm0 + ar1) * lda + k0 + ac1, lds_off_u32(&As[p][ar1][ac1]));
        async_b128(W + (bn0 + ar0) * (long)K + k0 + ac0, lds_off_u32(&Bs[p][ar0][ac0]));
        async_b128(W + (bn0 + ar1) * (long)K + k0 + ac1, lds_off_u32(&Bs[p][ar1][ac1]));
    };

    issue_tile(0, 0);
    int p = 0;

    for (int k0 = 0; k0 < K; k0 += BK) {
        const bool more = (k0 + BK) < K;
        if (more) {
            issue_tile(p ^ 1, k0 + BK);     // overlap next tile DMA with compute
            wait_async_le4();               // in-order: current tile's 4 landed
        } else {
            wait_async_0();
        }
        __syncthreads();                    // all waves' portions visible

        // ---- fragments: contiguous k order (k = 16*hi + s), same for A and B ----
        // A 16x32: lane -> row (lane&15); K = kb16..kb16+15 contiguous
        v16h afrag[2];
        #pragma unroll
        for (int i = 0; i < 2; ++i) {
            const int r = wm * 32 + i * 16 + l15;
            afrag[i] = *(const v16h*)&As[p][r][kb16];
        }
        // B 32x16: lane -> col (lane&15); K = kb16..kb16+15 contiguous
        // (Bmat[k][n] == W[n][k] -> contiguous 32B LDS read)
        v16h bfrag[4];
        #pragma unroll
        for (int j = 0; j < 4; ++j) {
            const int c = wn * 64 + j * 16 + l15;
            bfrag[j] = *(const v16h*)&Bs[p][c][kb16];
        }

        #pragma unroll
        for (int i = 0; i < 2; ++i)
            #pragma unroll
            for (int j = 0; j < 4; ++j)
                acc[i][j] = __builtin_amdgcn_wmma_f32_16x16x32_f16(
                    false, afrag[i], false, bfrag[j],
                    (short)0, acc[i][j], false, false);

        __syncthreads();                    // reads done before buffer reuse
        p ^= 1;
    }

    // ---- epilogue: bias (+tanh). C/D: lane0-15 N=lane,M=r; lane16-31 M=r+8
    const int mh = hi * 8;
    #pragma unroll
    for (int j = 0; j < 4; ++j) {
        const long gn = bn0 + wn * 64 + j * 16 + l15;
        const float bv = bias[gn];
        #pragma unroll
        for (int i = 0; i < 2; ++i) {
            #pragma unroll
            for (int r = 0; r < 8; ++r) {
                const long gm = bm0 + wm * 32 + i * 16 + mh + r;
                float v = acc[i][j][r] + bv;
                if (TANH) v = tanhf(v);
                if (OUT_F16) ((_Float16*)Out)[gm * ldo + gn] = (_Float16)v;
                else         ((float*)Out)[gm * ldo + gn] = v;
            }
        }
    }
}

// ---------------------------------------------------------------------------
// One-time f32 -> f16 conversions (hoisted out of the GEMM hot loops).
// ---------------------------------------------------------------------------
__global__ __launch_bounds__(256) void cvt_obs_kernel(const float* __restrict__ obs,
                                                      _Float16* __restrict__ X)
{
    const long i = (long)blockIdx.x * blockDim.x + threadIdx.x;  // over 64K x 512
    const long row = i >> 9, col = i & 511;
    X[i] = (_Float16)obs[row * OBS_STRIDE + col];
}

__global__ __launch_bounds__(256) void cvt_flat_kernel(const float* __restrict__ s,
                                                       _Float16* __restrict__ d, long n)
{
    const long i = (long)blockIdx.x * blockDim.x + threadIdx.x;
    if (i < n) d[i] = (_Float16)s[i];
}

// ---------------------------------------------------------------------------
// mask[m] = all(|gate| <= 0.1)
// ---------------------------------------------------------------------------
__global__ __launch_bounds__(256) void mask_kernel(const float* __restrict__ obs,
                                                   int* __restrict__ mask)
{
    const long m = (long)blockIdx.x * blockDim.x + threadIdx.x;
    if (m < M_ROWS) {
        const float* g = obs + m * OBS_STRIDE + IN_DIM;
        mask[m] = (fabsf(g[0]) <= 0.1f) & (fabsf(g[1]) <= 0.1f) & (fabsf(g[2]) <= 0.1f);
    }
}

// ---------------------------------------------------------------------------
// In-place masked softmax over 128 logits of d_out row; one wave32 per row.
// ---------------------------------------------------------------------------
__global__ __launch_bounds__(256) void softmax_mask_kernel(float* __restrict__ pi,
                                                           const int* __restrict__ mask,
                                                           int head)
{
    const int wave = threadIdx.x >> 5;
    const int lane = threadIdx.x & 31;
    const long m = (long)blockIdx.x * 8 + wave;
    float* row = pi + m * NA_DIM + head * 128;
    const bool keep = (mask[m] != 0) == (head == 0);

    float v[4];
    #pragma unroll
    for (int j = 0; j < 4; ++j) v[j] = row[lane + 32 * j];

    float mx = fmaxf(fmaxf(v[0], v[1]), fmaxf(v[2], v[3]));
    #pragma unroll
    for (int s = 16; s > 0; s >>= 1) mx = fmaxf(mx, __shfl_xor(mx, s, 32));

    float e[4], sum = 0.0f;
    #pragma unroll
    for (int j = 0; j < 4; ++j) { e[j] = __expf(v[j] - mx); sum += e[j]; }
    #pragma unroll
    for (int s = 16; s > 0; s >>= 1) sum += __shfl_xor(sum, s, 32);

    const float inv = 1.0f / sum;
    #pragma unroll
    for (int j = 0; j < 4; ++j) row[lane + 32 * j] = keep ? e[j] * inv : 0.0f;
}

// ---------------------------------------------------------------------------
// Value head: v[m] = dot(H2[m,:], w3) + b3   (N=1 -> wave dot product)
// ---------------------------------------------------------------------------
__global__ __launch_bounds__(256) void value_kernel(const _Float16* __restrict__ H2,
                                                    const float* __restrict__ w,
                                                    const float* __restrict__ b,
                                                    float* __restrict__ vout)
{
    const int wave = threadIdx.x >> 5;
    const int lane = threadIdx.x & 31;
    const long m = (long)blockIdx.x * 8 + wave;
    const _Float16* h = H2 + m * H_DIM;

    float s = 0.0f;
    for (int k = lane; k < H_DIM; k += 32) s += (float)h[k] * w[k];
    #pragma unroll
    for (int d = 16; d > 0; d >>= 1) s += __shfl_xor(s, d, 32);
    if (lane == 0) vout[m] = s + b[0];
}

// ---------------------------------------------------------------------------
extern "C" void kernel_launch(void* const* d_in, const int* in_sizes, int n_in,
                              void* d_out, int out_size, void* d_ws, size_t ws_size,
                              hipStream_t stream)
{
    (void)in_sizes; (void)n_in; (void)out_size; (void)ws_size;

    const float* obs = (const float*)d_in[0];
    // d_in[1]=prev_action, d_in[2]=prev_reward (unused by reference math)
    const float* w[3][3];   // [head][layer]  head: 0=pf, 1=pc, 2=value
    const float* bb[3][3];
    for (int h = 0; h < 3; ++h)
        for (int l = 0; l < 3; ++l) {
            w[h][l]  = (const float*)d_in[3 + h * 6 + l * 2];
            bb[h][l] = (const float*)d_in[3 + h * 6 + l * 2 + 1];
        }

    // ---- workspace layout (all 16B-aligned slabs) ----
    _Float16* X16 = (_Float16*)d_ws;                         // 64K x 512
    _Float16* H1  = X16 + (size_t)M_ROWS * IN_DIM;           // 64K x 1024
    _Float16* H2  = H1  + (size_t)M_ROWS * H_DIM;            // 64K x 1024
    _Float16* W1c = H2  + (size_t)M_ROWS * H_DIM;            // 1024 x 512
    _Float16* W2c = W1c + (size_t)H_DIM * IN_DIM;            // 1024 x 1024
    _Float16* W3c = W2c + (size_t)H_DIM * H_DIM;             // 128 x 1024
    int* mask     = (int*)(W3c + (size_t)128 * H_DIM);

    float* pi   = (float*)d_out;                             // [65536 x 256]
    float* vout = pi + (size_t)M_ROWS * NA_DIM;              // [65536]

    const dim3 blk(256);
    const dim3 gridH(M_ROWS / BM, H_DIM / BN);   // 512 x 8
    const dim3 gridP(M_ROWS / BM, 128 / BN);     // 512 x 1

    mask_kernel<<<M_ROWS / 256, blk, 0, stream>>>(obs, mask);
    cvt_obs_kernel<<<(M_ROWS * (long)IN_DIM) / 256, blk, 0, stream>>>(obs, X16);

    // ---- policy heads ----
    for (int head = 0; head < 2; ++head) {
        cvt_flat_kernel<<<(H_DIM * IN_DIM) / 256, blk, 0, stream>>>(w[head][0], W1c, (long)H_DIM * IN_DIM);
        cvt_flat_kernel<<<(H_DIM * H_DIM) / 256, blk, 0, stream>>>(w[head][1], W2c, (long)H_DIM * H_DIM);
        cvt_flat_kernel<<<(128 * H_DIM) / 256, blk, 0, stream>>>(w[head][2], W3c, (long)128 * H_DIM);

        gemm_wmma_async_kernel<true, true><<<gridH, blk, 0, stream>>>(
            X16, (long)IN_DIM, W1c, bb[head][0], H1, (long)H_DIM, IN_DIM);
        gemm_wmma_async_kernel<true, true><<<gridH, blk, 0, stream>>>(
            H1, (long)H_DIM, W2c, bb[head][1], H2, (long)H_DIM, H_DIM);
        gemm_wmma_async_kernel<false, false><<<gridP, blk, 0, stream>>>(
            H2, (long)H_DIM, W3c, bb[head][2], pi + head * 128, (long)NA_DIM, H_DIM);
        softmax_mask_kernel<<<M_ROWS / 8, blk, 0, stream>>>(pi, mask, head);
    }

    // ---- value head ----
    cvt_flat_kernel<<<(H_DIM * IN_DIM) / 256, blk, 0, stream>>>(w[2][0], W1c, (long)H_DIM * IN_DIM);
    cvt_flat_kernel<<<(H_DIM * H_DIM) / 256, blk, 0, stream>>>(w[2][1], W2c, (long)H_DIM * H_DIM);
    gemm_wmma_async_kernel<true, true><<<gridH, blk, 0, stream>>>(
        X16, (long)IN_DIM, W1c, bb[2][0], H1, (long)H_DIM, IN_DIM);
    gemm_wmma_async_kernel<true, true><<<gridH, blk, 0, stream>>>(
        H1, (long)H_DIM, W2c, bb[2][1], H2, (long)H_DIM, H_DIM);
    value_kernel<<<M_ROWS / 8, blk, 0, stream>>>(H2, w[2][2], bb[2][2], vout);
}
